// RationalBasicBlock_26654567039141
// MI455X (gfx1250) — matrix operand order repaired and checked by reading, not compiled, not run on hardware
//
#include <hip/hip_runtime.h>

typedef __attribute__((ext_vector_type(16))) _Float16 v16h;
typedef __attribute__((ext_vector_type(8)))  _Float16 v8h;
typedef __attribute__((ext_vector_type(8)))  float    v8f;

// Problem constants (reference: N=32, C=256, H=W=56)
namespace cfg {
constexpr int  N    = 32;
constexpr int  C    = 256;
constexpr int  H    = 56;
constexpr int  W    = 56;
constexpr int  HW   = H * W;                // 3136 (divisible by 64)
constexpr int  NPIX = N * HW;               // 100352 (divisible by 32)
constexpr long TOT  = (long)NPIX * C;       // 25,690,112 elements
constexpr int  KTOT = C * 9;                // 2304
constexpr int  KCH  = KTOT / 32;            // 72 K-chunks of 32
}

// ---------------------------------------------------------------------------
// Pack x: NCHW fp32 -> NHWC f16 via LDS 64x64 tile transpose.
// Both global sides fully coalesced; LDS padded (65) to avoid bank conflicts.
// Grid: (NPIX/64) * (C/64) blocks of 256 threads.
// ---------------------------------------------------------------------------
__global__ __launch_bounds__(256)
void pack_x_nhwc_f16(const float* __restrict__ x, _Float16* __restrict__ xh) {
    __shared__ float tile[64][65];
    const int pix0 = (blockIdx.x >> 2) * 64;         // 64 consecutive pixels
    const int c0   = (blockIdx.x & 3) * 64;          // 64 consecutive channels
    const int n    = pix0 / cfg::HW;                 // HW%64==0: no n crossing
    const int hw0  = pix0 - n * cfg::HW;
    const int t    = threadIdx.x;
    {   // coalesced NCHW reads (contiguous over hw)
        const int pl = t & 63, cr = t >> 6;
        const float* src = x + ((long)n * cfg::C + c0) * cfg::HW + hw0 + pl;
#pragma unroll
        for (int i = 0; i < 16; ++i) {
            int c = cr * 16 + i;
            tile[c][pl] = src[(long)c * cfg::HW];
        }
    }
    __syncthreads();
    {   // coalesced NHWC writes (contiguous over c)
        const int cl = t & 63, pr = t >> 6;
        _Float16* dst = xh + (long)pix0 * cfg::C + c0 + cl;
#pragma unroll
        for (int i = 0; i < 16; ++i) {
            int p = pr * 16 + i;
            dst[(long)p * cfg::C] = (_Float16)tile[cl][p];
        }
    }
}

// ---------------------------------------------------------------------------
// Pack weights [O,I,3,3] fp32 into WMMA B-fragment order, f16.
// K ordering: K = r*256 + c (r = kh*3+kw). Tile = (kc, ot), 512 halves/tile.
// Lane L of a tile holds column o = ot*16+(L&15), K = kc*32 + 16*(L>>4) + j.
// ---------------------------------------------------------------------------
__global__ __launch_bounds__(256)
void pack_w_frag(const float* __restrict__ w, _Float16* __restrict__ wp) {
    int idx = blockIdx.x * blockDim.x + threadIdx.x;          // 0..589823
    if (idx >= cfg::KTOT * cfg::C) return;
    int j    = idx & 15;
    int lane = (idx >> 4) & 31;
    int tile = idx >> 9;
    int ot   = tile & 15;
    int kc   = tile >> 4;                                     // 0..71
    int r    = kc >> 3;                                       // 0..8
    int chunk= kc & 7;
    int o    = ot * 16 + (lane & 15);
    int c    = chunk * 32 + ((lane >> 4) << 4) + j;
    wp[idx] = (_Float16)w[((long)o * cfg::C + c) * 9 + r];
}

// ---------------------------------------------------------------------------
// Implicit-GEMM 3x3 conv via V_WMMA_F32_16X16X32_F16.
// Block = 8 waves: 2 pixel-tiles (16 px) x 4 channel-groups (64 oc).
// Per K-chunk the block's 16 KB weight slab is DMA'd to LDS once with
// global_load_async_to_lds_b128 (ASYNCcnt), double-buffered so DMA(kc+1)
// overlaps WMMA(kc). A fragments stream straight from global (the real HBM
// traffic), B fragments come from LDS (ds_load_b128).
// ---------------------------------------------------------------------------
__global__ __launch_bounds__(256)
void conv3x3_wmma(const _Float16* __restrict__ A,   // NHWC f16 activations
                  const _Float16* __restrict__ Bp,  // packed weight fragments
                  float* __restrict__ Y) {          // NHWC f32 output
    constexpr int C = cfg::C, H = cfg::H, W = cfg::W;
    __shared__ _Float16 shB[2][8192];               // 2 x 16 KB K-chunk slabs

    const int tid   = threadIdx.x;
    const int lane  = tid & 31;
    const int wv    = tid >> 5;                     // 0..7
    const int mtile = blockIdx.x * 2 + (wv >> 2);   // 16-pixel tile
    const int og    = wv & 3;                       // 64-channel group

    const unsigned shBase = (unsigned)(size_t)&shB[0][0];  // LDS byte offset

    const int mrow = lane & 15;
    const int grp  = lane >> 4;                     // K-half this lane supplies
    const int m    = mtile * 16 + mrow;
    const int n    = m / cfg::HW;
    const int rem  = m - n * cfg::HW;
    const int h    = rem / W;
    const int w    = rem - h * W;

    // DMA one 16KB weight K-chunk into LDS buffer `buf` (4 b128 per thread).
    auto issueB = [&](int kc, int buf) {
        unsigned lo = shBase + (unsigned)buf * 16384u + (unsigned)tid * 16u;
        const char* g = (const char*)Bp + (size_t)kc * 16384 + (size_t)tid * 16;
#pragma unroll
        for (int i = 0; i < 4; ++i) {
            asm volatile("global_load_async_to_lds_b128 %0, %1, off"
                         :: "v"(lo + (unsigned)i * 4096u), "v"(g + i * 4096)
                         : "memory");
        }
    };

    v8f acc0 = {}, acc1 = {}, acc2 = {}, acc3 = {};

    issueB(0, 0);
    for (int kc = 0; kc < cfg::KCH; ++kc) {
        if (kc + 1 < cfg::KCH) {
            issueB(kc + 1, (kc + 1) & 1);           // overlap DMA with compute
            asm volatile("s_wait_asynccnt 0x4" ::: "memory");  // cur slab done
        } else {
            asm volatile("s_wait_asynccnt 0x0" ::: "memory");
        }
        __syncthreads();                            // slab visible to all waves

        // A fragment: 16 halves/lane = two contiguous 16B loads (or 0 pad)
        const int  r  = kc >> 3, ch = kc & 7;
        const int  hh = h + r / 3 - 1, ww = w + r % 3 - 1;
        const bool valid = (hh >= 0) && (hh < H) && (ww >= 0) && (ww < W);
        v16h afrag = {};
        if (valid) {
            const _Float16* p =
                A + (long)((n * H + hh) * W + ww) * C + ch * 32 + grp * 16;
            if (ch == 0) __builtin_prefetch(p + 128, 0, 1);
            v8h lo = *(const v8h*)(p);
            v8h hi = *(const v8h*)(p + 8);
            afrag = __builtin_shufflevector(lo, hi,
                     0,1,2,3,4,5,6,7,8,9,10,11,12,13,14,15);
        }

        // B fragments from LDS (16B-aligned -> ds_load_b128)
        const _Float16* bb = &shB[kc & 1][og * 2048 + lane * 16];
#define B_FRAG(T)                                                         \
        v8h bl##T = *(const v8h*)(bb + (T) * 512);                        \
        v8h bh##T = *(const v8h*)(bb + (T) * 512 + 8);                    \
        v16h bf##T = __builtin_shufflevector(bl##T, bh##T,                \
                     0,1,2,3,4,5,6,7,8,9,10,11,12,13,14,15)
        B_FRAG(0); B_FRAG(1); B_FRAG(2); B_FRAG(3);
#undef B_FRAG
        acc0 = __builtin_amdgcn_wmma_f32_16x16x32_f16(
                   false, afrag, false, bf0, (short)0, acc0, false, false);
        acc1 = __builtin_amdgcn_wmma_f32_16x16x32_f16(
                   false, afrag, false, bf1, (short)0, acc1, false, false);
        acc2 = __builtin_amdgcn_wmma_f32_16x16x32_f16(
                   false, afrag, false, bf2, (short)0, acc2, false, false);
        acc3 = __builtin_amdgcn_wmma_f32_16x16x32_f16(
                   false, afrag, false, bf3, (short)0, acc3, false, false);

        __syncthreads();                            // slab free for next DMA
    }

    // C/D layout: lane L -> col (L&15); VGPR j -> row j + 8*(L>>4)
    const int ncol = lane & 15;
    float* yb = Y + (long)(mtile * 16 + 8 * grp) * C + og * 64 + ncol;
#pragma unroll
    for (int j = 0; j < 8; ++j) {
        yb[(long)j * C +  0] = acc0[j];
        yb[(long)j * C + 16] = acc1[j];
        yb[(long)j * C + 32] = acc2[j];
        yb[(long)j * C + 48] = acc3[j];
    }
}

// ---------------------------------------------------------------------------
// BatchNorm helpers
// ---------------------------------------------------------------------------
__global__ __launch_bounds__(512)
void zero_stats(float* __restrict__ s) {         // sums[256] ++ sqs[256]
    if (threadIdx.x < 512) s[threadIdx.x] = 0.0f;
}

__global__ __launch_bounds__(256)
void bn_stats(const float* __restrict__ Y, float* __restrict__ sums) {
    const int c = threadIdx.x;                   // channel (coalesced reads)
    float s = 0.0f, q = 0.0f;
    for (int m = blockIdx.x; m < cfg::NPIX; m += gridDim.x) {
        float v = Y[(long)m * cfg::C + c];
        s += v;
        q = fmaf(v, v, q);
    }
    atomicAdd(&sums[c], s);                      // global_atomic_add_f32
    atomicAdd(&sums[cfg::C + c], q);
}

__global__ __launch_bounds__(256)
void bn_finalize(const float* __restrict__ sums,
                 const float* __restrict__ gamma, const float* __restrict__ beta,
                 float* __restrict__ scale_shift) {
    const int c = threadIdx.x;
    const float inv = 1.0f / (float)cfg::NPIX;
    float mean = sums[c] * inv;
    float var  = fmaf(-mean, mean, sums[cfg::C + c] * inv);
    float sc   = gamma[c] * rsqrtf(var + 1e-5f);
    scale_shift[c]          = sc;
    scale_shift[cfg::C + c] = fmaf(-mean, sc, beta[c]);
}

__device__ __forceinline__ float rational_f(float x,
                                            const float* __restrict__ a,
                                            const float* __restrict__ b) {
    float P = fmaf(x, fmaf(x, fmaf(x, fmaf(x, fmaf(x, a[5], a[4]), a[3]),
                                   a[2]), a[1]), a[0]);
    float Q = 1.0f + fabsf(x * fmaf(x, fmaf(x, fmaf(x, b[3], b[2]), b[1]),
                                    b[0]));
    return P / Q;
}

// BN apply + rational + convert to f16 NHWC (input for conv2); fully coalesced
__global__ __launch_bounds__(256)
void bn_act_pack(const float* __restrict__ Y,
                 const float* __restrict__ scale_shift,
                 const float* __restrict__ ra, const float* __restrict__ rb,
                 _Float16* __restrict__ acth) {
    long idx = (long)blockIdx.x * blockDim.x + threadIdx.x;
    if (idx >= cfg::TOT) return;
    int c = (int)(idx & (cfg::C - 1));
    float v = fmaf(Y[idx], scale_shift[c], scale_shift[cfg::C + c]);
    acth[idx] = (_Float16)rational_f(v, ra, rb);
}

// BN apply + residual + rational, NHWC f32 -> NCHW f32 output.
// LDS 64x64 tile transpose so both global sides are coalesced.
__global__ __launch_bounds__(256)
void bn_res_act_out(const float* __restrict__ Y,
                    const float* __restrict__ x,
                    const float* __restrict__ scale_shift,
                    const float* __restrict__ ra, const float* __restrict__ rb,
                    float* __restrict__ out) {
    __shared__ float tile[64][65];
    const int pix0 = (blockIdx.x >> 2) * 64;
    const int c0   = (blockIdx.x & 3) * 64;
    const int n    = pix0 / cfg::HW;
    const int hw0  = pix0 - n * cfg::HW;
    const int t    = threadIdx.x;
    {   // coalesced NHWC reads of conv output (contiguous over c)
        const int cl = t & 63, pr = t >> 6;
        const float* src = Y + (long)pix0 * cfg::C + c0 + cl;
#pragma unroll
        for (int i = 0; i < 16; ++i) {
            int p = pr * 16 + i;
            tile[p][cl] = src[(long)p * cfg::C];
        }
    }
    __syncthreads();
    {   // coalesced NCHW side: BN + residual + rational + store
        const int pl = t & 63, cr = t >> 6;
#pragma unroll
        for (int i = 0; i < 16; ++i) {
            int  c   = cr * 16 + i;
            long idx = ((long)n * cfg::C + c0 + c) * cfg::HW + hw0 + pl;
            float v  = fmaf(tile[pl][c], scale_shift[c0 + c],
                            scale_shift[cfg::C + c0 + c]) + x[idx];
            out[idx] = rational_f(v, ra, rb);
        }
    }
}

// ---------------------------------------------------------------------------
// Launch
// ---------------------------------------------------------------------------
extern "C" void kernel_launch(void* const* d_in, const int* in_sizes, int n_in,
                              void* d_out, int out_size, void* d_ws,
                              size_t ws_size, hipStream_t stream) {
    const float* x  = (const float*)d_in[0];
    const float* w1 = (const float*)d_in[1];
    const float* g1 = (const float*)d_in[2];
    const float* b1 = (const float*)d_in[3];
    const float* w2 = (const float*)d_in[4];
    const float* g2 = (const float*)d_in[5];
    const float* b2 = (const float*)d_in[6];
    const float* ra = (const float*)d_in[7];
    const float* rb = (const float*)d_in[8];
    float* out = (float*)d_out;

    // Workspace layout (~156.5 MB total)
    char* ws = (char*)d_ws;
    const size_t XH_BYTES = (size_t)cfg::TOT * 2;           //  51,380,224
    const size_t Y_BYTES  = (size_t)cfg::TOT * 4;           // 102,760,448
    const size_t WP_BYTES = (size_t)cfg::KTOT * cfg::C * 2; //   1,179,648
    _Float16* xh   = (_Float16*)ws;                 // x f16 NHWC (reused: act1)
    float*    Y    = (float*)(ws + XH_BYTES);       // conv out NHWC (reused)
    _Float16* w1p  = (_Float16*)(ws + XH_BYTES + Y_BYTES);
    _Float16* w2p  = (_Float16*)(ws + XH_BYTES + Y_BYTES + WP_BYTES);
    float*    sums = (float*)(ws + XH_BYTES + Y_BYTES + 2 * WP_BYTES);
    float*    ss1  = sums + 512;                    // scale/shift BN1
    float*    ss2  = ss1 + 512;                     // scale/shift BN2

    const int ELEM_BLOCKS = (int)((cfg::TOT + 255) / 256);     // 100352
    const int TILE_BLOCKS = (cfg::NPIX / 64) * (cfg::C / 64);  // 6272
    const int CONV_BLOCKS = (cfg::NPIX / 16) / 2;              // 3136
    const int PW_BLOCKS   = (cfg::KTOT * cfg::C + 255) / 256;

    // Stage 0: pack activations + both weight tensors
    pack_x_nhwc_f16<<<TILE_BLOCKS, 256, 0, stream>>>(x, xh);
    pack_w_frag<<<PW_BLOCKS, 256, 0, stream>>>(w1, w1p);
    pack_w_frag<<<PW_BLOCKS, 256, 0, stream>>>(w2, w2p);

    // Stage 1: conv1 -> BN1 stats -> BN1+rational -> act1 (f16, aliases xh)
    conv3x3_wmma<<<CONV_BLOCKS, 256, 0, stream>>>(xh, w1p, Y);
    zero_stats<<<1, 512, 0, stream>>>(sums);
    bn_stats<<<448, 256, 0, stream>>>(Y, sums);
    bn_finalize<<<1, 256, 0, stream>>>(sums, g1, b1, ss1);
    bn_act_pack<<<ELEM_BLOCKS, 256, 0, stream>>>(Y, ss1, ra, rb, xh);

    // Stage 2: conv2 (output aliases Y) -> BN2 stats -> BN2+res+rational
    conv3x3_wmma<<<CONV_BLOCKS, 256, 0, stream>>>(xh, w2p, Y);
    zero_stats<<<1, 512, 0, stream>>>(sums);
    bn_stats<<<448, 256, 0, stream>>>(Y, sums);
    bn_finalize<<<1, 256, 0, stream>>>(sums, g2, b2, ss2);
    bn_res_act_out<<<TILE_BLOCKS, 256, 0, stream>>>(Y, x, ss2, ra, rb, out);
}
